// GELU46_17566416240666
// MI455X (gfx1250) — matrix operand order repaired and checked by reading, not compiled, not run on hardware
//
#include <hip/hip_runtime.h>
#include <stdint.h>

// CDNA5 / gfx1250 flash-attention implementation of the gelu + cosine-sim
// causal mixing reference. wave32, v_wmma_f32_16x16x32_f16 for both matmuls,
// double-buffered GLOBAL_LOAD_ASYNC_TO_LDS tile staging (ASYNCcnt).

typedef __attribute__((ext_vector_type(16))) _Float16 v16h;
typedef __attribute__((ext_vector_type(8)))  _Float16 v8h;
typedef __attribute__((ext_vector_type(4)))  _Float16 v4h;
typedef __attribute__((ext_vector_type(8)))  float    v8f;

#define T_SEQ 8192
#define D_DIM 128
#define BQ    64              // query rows per block (16 per wave, 4 waves)
#define BK    64              // key rows per LDS tile
#define NKT   (T_SEQ / BK)    // 128 key tiles per batch
#define NWAVE 4
#define KSTR  144             // K tile row stride (halves): 128+16 pad
#define VSTR  72              // Vt tile row stride (halves): 64+8 pad
#define TILE_BYTES 18432      // 64*144*2 == 128*72*2
#define CHUNKS_PER_THREAD 9   // 18432 / (128 threads * 16B)

__device__ __forceinline__ float softplusf(float x) {
  return (x > 20.f) ? x : log1pf(expf(x));
}
__device__ __forceinline__ float gelu_tanh(float x) {
  const float c = 0.7978845608028654f;  // sqrt(2/pi)
  float x3 = x * x * x;
  return 0.5f * x * (1.f + tanhf(c * (x + 0.044715f * x3)));
}
__device__ __forceinline__ float rmax16(float v) {
#pragma unroll
  for (int m = 1; m < 16; m <<= 1) v = fmaxf(v, __shfl_xor(v, m, 32));
  return v;
}
__device__ __forceinline__ float rsum16(float v) {
#pragma unroll
  for (int m = 1; m < 16; m <<= 1) v += __shfl_xor(v, m, 32);
  return v;
}
__device__ __forceinline__ v16h cat16(v8h lo, v8h hi) {
  return __builtin_shufflevector(lo, hi, 0, 1, 2, 3, 4, 5, 6, 7,
                                 8, 9, 10, 11, 12, 13, 14, 15);
}
// Async copy of one 16B chunk, global -> LDS (tracked by ASYNCcnt).
__device__ __forceinline__ void async_b128(uint32_t lds_off, const void* gptr) {
  asm volatile("global_load_async_to_lds_b128 %0, %1, off"
               :
               : "v"(lds_off), "v"(gptr)
               : "memory");
}

// ---------------------------------------------------------------------------
// Kernel 1: out = gelu(x); on = out/max(||out||,1e-6).
// Writes out (f32) to d_out; on (f16) into padded K tiles [b][kt][64][KSTR];
// gelu (f16) transposed into padded Vt tiles [b][kt][128][VSTR].
// One wave per row: 32 lanes x 4 floats = 128 = D.
// ---------------------------------------------------------------------------
__global__ __launch_bounds__(128) void gelu_norm_kernel(
    const float* __restrict__ x, float* __restrict__ outf,
    _Float16* __restrict__ kt_t, _Float16* __restrict__ vt_t, int nrows) {
  const int wave = threadIdx.x >> 5;
  const int lane = threadIdx.x & 31;
  const int row = blockIdx.x * 4 + wave;
  if (row >= nrows) return;
  const int b = row / T_SEQ, t = row % T_SEQ;
  const int kt = t >> 6, kin = t & 63;

  const float4 xv = *(const float4*)(x + (size_t)row * D_DIM + lane * 4);
  float g0 = gelu_tanh(xv.x), g1 = gelu_tanh(xv.y);
  float g2 = gelu_tanh(xv.z), g3 = gelu_tanh(xv.w);

  float ss = g0 * g0 + g1 * g1 + g2 * g2 + g3 * g3;
#pragma unroll
  for (int m = 1; m < 32; m <<= 1) ss += __shfl_xor(ss, m, 32);
  const float inv = 1.f / fmaxf(sqrtf(ss), 1e-6f);

  *(float4*)(outf + (size_t)row * D_DIM + lane * 4) = make_float4(g0, g1, g2, g3);

  // normalized row into padded K tile (row-major within tile)
  _Float16* kd = kt_t + (((size_t)b * NKT + kt) * BK + kin) * KSTR + lane * 4;
  v4h nv = {(_Float16)(g0 * inv), (_Float16)(g1 * inv),
            (_Float16)(g2 * inv), (_Float16)(g3 * inv)};
  *(v4h*)kd = nv;

  // gelu row scattered into transposed Vt tile: Vt[d][kin]
  _Float16* vb = vt_t + ((size_t)b * NKT + kt) * (D_DIM * VSTR) + kin;
  vb[(size_t)(lane * 4 + 0) * VSTR] = (_Float16)g0;
  vb[(size_t)(lane * 4 + 1) * VSTR] = (_Float16)g1;
  vb[(size_t)(lane * 4 + 2) * VSTR] = (_Float16)g2;
  vb[(size_t)(lane * 4 + 3) * VSTR] = (_Float16)g3;
}

// ---------------------------------------------------------------------------
// Kernel 2: causal flash attention with online softmax + fused epilogue:
//   d_out = out + alpha*(out - softmax(beta*Q.K^T/sqrt(D)) . V)
// 4 waves/block, 64 queries/block (16/wave), 64-key double-buffered tiles
// staged with global_load_async_to_lds_b128.
// ---------------------------------------------------------------------------
__global__ __launch_bounds__(128) void flash_mix_kernel(
    const _Float16* __restrict__ kt_t, const _Float16* __restrict__ vt_t,
    float* __restrict__ outf, const float* __restrict__ log_beta_raw,
    const float* __restrict__ log_alpha_raw) {
  __shared__ _Float16 Ks[2][BK][KSTR];        // K tiles  [key][dim] (padded)
  __shared__ _Float16 Vts[2][D_DIM][VSTR];    // V tiles  [dim][key] (padded)
  __shared__ _Float16 Ps[NWAVE][16][VSTR];    // per-wave P staging (C->A relayout)

  const int tid = threadIdx.x;
  const int wave = tid >> 5, lane = tid & 31;
  const int ln16 = lane & 15, sel = lane >> 4;

  const int nqt = T_SEQ / BQ;
  const int b = blockIdx.x / nqt;
  const int qt = blockIdx.x % nqt;
  const int qbase = qt * BQ;

  const float beta = fminf(softplusf(log_beta_raw[0]), 5.f) + 0.5f;
  const float alpha = softplusf(log_alpha_raw[0]);
  // beta/sqrt(128) * log2(e): softmax done in exp2 domain (identical result)
  const float scale2 = beta * 0.08838834764831845f * 1.4426950408889634f;

  const _Float16* ktiles = kt_t + (size_t)b * NKT * (BK * KSTR);
  const _Float16* vtiles = vt_t + (size_t)b * NKT * (D_DIM * VSTR);
  const uint32_t ks_off[2] = {(uint32_t)(uintptr_t)&Ks[0][0][0],
                              (uint32_t)(uintptr_t)&Ks[1][0][0]};
  const uint32_t vs_off[2] = {(uint32_t)(uintptr_t)&Vts[0][0][0],
                              (uint32_t)(uintptr_t)&Vts[1][0][0]};

  // --- Q fragments, A layout (ISA 7.12.2): lane row = ln16,
  //     halves {sel*8..sel*8+7} and {16+sel*8..}, 4 chunks of K=32.
  const _Float16* qp = ktiles + ((size_t)qt * BK + wave * 16 + ln16) * KSTR;
  v16h qa[4];
#pragma unroll
  for (int c = 0; c < 4; ++c) {
    v8h lo = *(const v8h*)(qp + c * 32 + sel * 8);
    v8h hi = *(const v8h*)(qp + c * 32 + 16 + sel * 8);
    qa[c] = cat16(lo, hi);
  }

  v8f Oacc[8];
#pragma unroll
  for (int i = 0; i < 8; ++i) Oacc[i] = (v8f){};
  float mrow[8], lrow[8];
#pragma unroll
  for (int r = 0; r < 8; ++r) { mrow[r] = -1e30f; lrow[r] = 0.f; }

  // Issue async copy of tile 0 into buffer 0 (18 b128 ops per thread).
  {
    const char* ksrc = (const char*)(ktiles) + (size_t)0;
    const char* vsrc = (const char*)(vtiles) + (size_t)0;
#pragma unroll
    for (int i = 0; i < CHUNKS_PER_THREAD; ++i) {
      const uint32_t off = (uint32_t)(i * 128 + tid) * 16u;
      async_b128(ks_off[0] + off, ksrc + off);
    }
#pragma unroll
    for (int i = 0; i < CHUNKS_PER_THREAD; ++i) {
      const uint32_t off = (uint32_t)(i * 128 + tid) * 16u;
      async_b128(vs_off[0] + off, vsrc + off);
    }
  }

  for (int kt = 0; kt <= qt; ++kt) {
    const int cur = kt & 1;

    if (kt < qt) {
      // Issue next tile's copy into the other buffer (reads of it finished
      // at the previous iteration's trailing barrier), then wait for the
      // current tile: 18 newer ops outstanding, async ops retire in order.
      const int nxt = cur ^ 1;
      const char* ksrc = (const char*)ktiles + (size_t)(kt + 1) * TILE_BYTES;
      const char* vsrc = (const char*)vtiles + (size_t)(kt + 1) * TILE_BYTES;
#pragma unroll
      for (int i = 0; i < CHUNKS_PER_THREAD; ++i) {
        const uint32_t off = (uint32_t)(i * 128 + tid) * 16u;
        async_b128(ks_off[nxt] + off, ksrc + off);
      }
#pragma unroll
      for (int i = 0; i < CHUNKS_PER_THREAD; ++i) {
        const uint32_t off = (uint32_t)(i * 128 + tid) * 16u;
        async_b128(vs_off[nxt] + off, vsrc + off);
      }
      asm volatile("s_wait_asynccnt 0x12" ::: "memory");  // <= 18
    } else {
      asm volatile("s_wait_asynccnt 0x0" ::: "memory");
    }
    __syncthreads();  // current tile visible to all waves

    // --- S = Q K^T : 4 column tiles of 16 keys, K accumulated over 4x32.
    v8f s[4];
#pragma unroll
    for (int ct = 0; ct < 4; ++ct) {
      v8f acc = (v8f){};
#pragma unroll
      for (int c = 0; c < 4; ++c) {
        // B fragment: lane col = key ln16, dims {sel*8..sel*8+7} and {16+sel*8..}.
        const _Float16* bp = &Ks[cur][ct * 16 + ln16][c * 32 + sel * 8];
        v16h bf = cat16(*(const v8h*)bp, *(const v8h*)(bp + 16));
        acc = __builtin_amdgcn_wmma_f32_16x16x32_f16(false, qa[c], false, bf,
                                                     (short)0, acc, false, false);
      }
      s[ct] = acc;
    }

    // --- scale (exp2 domain) + causal mask
    // (C layout: row = r + sel*8, col = ct*16+ln16)
    const bool diag = (kt == qt);
#pragma unroll
    for (int ct = 0; ct < 4; ++ct) {
#pragma unroll
      for (int r = 0; r < 8; ++r) {
        float v = s[ct][r] * scale2;
        if (diag && (ct * 16 + ln16) > (wave * 16 + r + sel * 8)) v = -1e30f;
        s[ct][r] = v;
      }
    }

    // --- online softmax update (base-2)
    float f[8];
#pragma unroll
    for (int r = 0; r < 8; ++r) {
      float t = fmaxf(fmaxf(s[0][r], s[1][r]), fmaxf(s[2][r], s[3][r]));
      t = rmax16(t);
      const float mn = fmaxf(mrow[r], t);
      f[r] = exp2f(mrow[r] - mn);
      mrow[r] = mn;
    }
#pragma unroll
    for (int r = 0; r < 8; ++r) {
      float rs = 0.f;
#pragma unroll
      for (int ct = 0; ct < 4; ++ct) {
        const float p = exp2f(s[ct][r] - mrow[r]);
        rs += p;
        Ps[wave][r + sel * 8][ct * 16 + ln16] = (_Float16)p;
      }
      lrow[r] = lrow[r] * f[r] + rsum16(rs);
#pragma unroll
      for (int dt = 0; dt < 8; ++dt) Oacc[dt][r] *= f[r];
    }
    __syncthreads();  // Ps visible across lanes

    // --- O += P.V : 8 output dim tiles, K(=keys) accumulated over 2x32.
#pragma unroll
    for (int ch = 0; ch < 2; ++ch) {
      const _Float16* ap = &Ps[wave][ln16][ch * 32 + sel * 8];
      v16h af = cat16(*(const v8h*)ap, *(const v8h*)(ap + 16));
#pragma unroll
      for (int dt = 0; dt < 8; ++dt) {
        const _Float16* bp = &Vts[cur][dt * 16 + ln16][ch * 32 + sel * 8];
        v16h bf = cat16(*(const v8h*)bp, *(const v8h*)(bp + 16));
        Oacc[dt] = __builtin_amdgcn_wmma_f32_16x16x32_f16(
            false, af, false, bf, (short)0, Oacc[dt], false, false);
      }
    }
    __syncthreads();  // all reads of buf[cur] done before it is re-filled
  }

  // --- epilogue: result = out + alpha*(out - O/l)
#pragma unroll
  for (int r = 0; r < 8; ++r) {
    const float invl = 1.f / lrow[r];
    const int q = qbase + wave * 16 + r + sel * 8;
    float* op = outf + ((size_t)b * T_SEQ + q) * D_DIM;
#pragma unroll
    for (int dt = 0; dt < 8; ++dt) {
      const int d = dt * 16 + ln16;
      const float mu = Oacc[dt][r] * invl;
      const float o = op[d];
      op[d] = o + alpha * (o - mu);
    }
  }
}

// ---------------------------------------------------------------------------
extern "C" void kernel_launch(void* const* d_in, const int* in_sizes, int n_in,
                              void* d_out, int out_size, void* d_ws,
                              size_t ws_size, hipStream_t stream) {
  const float* x = (const float*)d_in[0];
  const float* log_beta = (const float*)d_in[1];
  const float* log_alpha = (const float*)d_in[2];
  float* outf = (float*)d_out;

  const int B = in_sizes[0] / (T_SEQ * D_DIM);  // = 2
  // workspace: padded K tiles then padded Vt tiles (TILE_BYTES each)
  _Float16* kt_t = (_Float16*)d_ws;
  _Float16* vt_t = kt_t + (size_t)B * NKT * (BK * KSTR);

  const int nrows = B * T_SEQ;
  hipLaunchKernelGGL(gelu_norm_kernel, dim3(nrows / 4), dim3(128), 0, stream,
                     x, outf, kt_t, vt_t, nrows);
  hipLaunchKernelGGL(flash_mix_kernel, dim3(B * (T_SEQ / BQ)), dim3(128), 0,
                     stream, kt_t, vt_t, outf, log_beta, log_alpha);
}
